// FlaxBloomAttention_24756191494656
// MI455X (gfx1250) — compile-verified, matrix operand order checked by reading, not run on hardware
//
#include <hip/hip_runtime.h>
#include <hip/hip_bf16.h>

// ---------------------------------------------------------------------------
// Bloom attention block on MI455X (gfx1250), wave32 + WMMA bf16 + async LDS.
// B=2, S=2048, H=2048, NH=16, HD=128.
// ---------------------------------------------------------------------------

#define B_   2
#define S_   2048
#define H_   2048
#define NH_  16
#define HD_  128
#define H3_  (3 * H_)   // 6144

typedef __attribute__((ext_vector_type(16))) __bf16        v16bf;
typedef __attribute__((ext_vector_type(8)))  float         v8f;
typedef __attribute__((ext_vector_type(8)))  unsigned int  v8u;

__device__ __forceinline__ unsigned short f32_to_bf16(float f) {
    union { float f; unsigned int u; } v; v.f = f;
    unsigned int u = v.u;
    unsigned int r = u + 0x7FFFu + ((u >> 16) & 1u);   // round-to-nearest-even
    return (unsigned short)(r >> 16);
}

// LDS byte offset of a shared-memory generic pointer (flat LDS aperture keeps
// the wave-relative LDS address in addr[31:0] per ISA 10.2).
__device__ __forceinline__ unsigned int lds_offset(const void* p) {
    return (unsigned int)(unsigned long long)p;
}

// CDNA5 async copy: 16B global -> LDS per lane, tracked by ASYNCcnt.
__device__ __forceinline__ void async_copy_b128(unsigned int ldsoff, const void* g) {
    asm volatile("global_load_async_to_lds_b128 %0, %1, off"
                 :: "v"(ldsoff), "v"(g) : "memory");
}
__device__ __forceinline__ void wait_async0() {
    asm volatile("s_wait_asynccnt 0x0" ::: "memory");
}

// Load one 16x32 bf16 WMMA A/B fragment from a row-major tile.
// ISA 16-bit A layout: lane L holds row (L&15); lanes 0-15 carry K={0..7,16..23},
// lanes 16-31 carry K={8..15,24..31}.  Two 16B LDS loads per lane.
__device__ __forceinline__ v16bf load_frag16(const unsigned short* tile, int stride,
                                             int koff, int lane) {
    int r  = lane & 15;
    int kb = ((lane >> 4) & 1) * 8;
    const unsigned short* row = tile + (size_t)r * stride + koff + kb;
    uint4 lo = *(const uint4*)(row);
    uint4 hi = *(const uint4*)(row + 16);
    v8u v;
    v[0] = lo.x; v[1] = lo.y; v[2] = lo.z; v[3] = lo.w;
    v[4] = hi.x; v[5] = hi.y; v[6] = hi.z; v[7] = hi.w;
    return __builtin_bit_cast(v16bf, v);
}

__device__ __forceinline__ v8f wmma_bf16(v16bf a, v16bf b, v8f c) {
    return __builtin_amdgcn_wmma_f32_16x16x32_bf16(
        false, a, false, b, (short)0, c, false, false);
}

// ---------------------------------------------------------------------------
// fp32 -> bf16 conversion (grid-stride)
// ---------------------------------------------------------------------------
__global__ void convert_bf16_kernel(const float* __restrict__ in,
                                    unsigned short* __restrict__ out, size_t n) {
    size_t i = (size_t)blockIdx.x * blockDim.x + threadIdx.x;
    size_t stride = (size_t)gridDim.x * blockDim.x;
    for (; i < n; i += stride) out[i] = f32_to_bf16(in[i]);
}

// ---------------------------------------------------------------------------
// Tiled bf16 GEMM:  C[M,N] = A[M,K] * B[K,N] (+bias[N]) (+resid[M,N])
// Block tile 128x128, K-step 32, 256 threads = 8 waves (4x2 wave grid),
// each wave computes 32x64 (2x4 WMMA tiles).  Double-buffered LDS with
// async global->LDS staging of A overlapping the WMMA compute of the
// previous K-step.  All dims are multiples of tile sizes for our shapes.
// ---------------------------------------------------------------------------
#define STA 48   // 32 + 16 pad elems; 96 bytes/row, 16B aligned
#define STB 48

template <bool OUT_BF16>
__global__ __launch_bounds__(256)
void gemm_bf16_kernel(const unsigned short* __restrict__ A,
                      const unsigned short* __restrict__ Bm,
                      const float* __restrict__ bias,
                      const float* __restrict__ resid,
                      void* __restrict__ out,
                      int M, int N, int K) {
    __shared__ unsigned short As[2][128 * STA];
    __shared__ unsigned short Bt[2][128 * STB];   // transposed: [n][k]

    const int t      = threadIdx.x;
    const int lane   = t & 31;
    const int wave   = t >> 5;
    const int wave_m = wave >> 1;   // 0..3
    const int wave_n = wave & 1;    // 0..1
    const int m0     = blockIdx.y * 128;
    const int n0     = blockIdx.x * 128;

    // Per-thread staging coordinates (2 x 16B chunks each for A and B).
    const int a_row[2] = { (t + 0)   >> 2, (t + 256) >> 2 };
    const int a_c8 [2] = { (t & 3) * 8,    (t & 3) * 8    };
    const int b_kr [2] = { (t + 0)   >> 4, (t + 256) >> 4 };
    const int b_c8 [2] = { (t & 15) * 8,   (t & 15) * 8   };

    v8f acc[2][4];
#pragma unroll
    for (int i = 0; i < 2; ++i)
#pragma unroll
        for (int j = 0; j < 4; ++j) acc[i][j] = (v8f)0.0f;

    // ---- prologue: stage K-tile 0 into buffer 0 ----
#pragma unroll
    for (int i = 0; i < 2; ++i)
        async_copy_b128(lds_offset(&As[0][a_row[i] * STA + a_c8[i]]),
                        A + (size_t)(m0 + a_row[i]) * K + a_c8[i]);
    {
        union { uint4 v; unsigned short s[8]; } u[2];
#pragma unroll
        for (int i = 0; i < 2; ++i)
            u[i].v = *(const uint4*)(Bm + (size_t)b_kr[i] * N + n0 + b_c8[i]);
#pragma unroll
        for (int i = 0; i < 2; ++i)
#pragma unroll
            for (int q = 0; q < 8; ++q)
                Bt[0][(b_c8[i] + q) * STB + b_kr[i]] = u[i].s[q];
    }
    wait_async0();
    __syncthreads();

    const int nsteps = K / 32;
    for (int step = 0; step < nsteps; ++step) {
        const int cur = step & 1;
        const int nxt = cur ^ 1;
        const bool has_next = (step + 1) < nsteps;

        // ---- issue async A(next) + pull B(next) into regs (overlaps WMMA) ---
        union { uint4 v; unsigned short s[8]; } breg[2];
        if (has_next) {
            const int k1 = (step + 1) * 32;
#pragma unroll
            for (int i = 0; i < 2; ++i)
                async_copy_b128(lds_offset(&As[nxt][a_row[i] * STA + a_c8[i]]),
                                A + (size_t)(m0 + a_row[i]) * K + k1 + a_c8[i]);
#pragma unroll
            for (int i = 0; i < 2; ++i)
                breg[i].v = *(const uint4*)(Bm + (size_t)(k1 + b_kr[i]) * N +
                                            n0 + b_c8[i]);
        }

        // ---- compute on current buffers ----
        v16bf afr[2], bfr[4];
#pragma unroll
        for (int i = 0; i < 2; ++i)
            afr[i] = load_frag16(&As[cur][(wave_m * 32 + i * 16) * STA], STA, 0, lane);
#pragma unroll
        for (int j = 0; j < 4; ++j)
            bfr[j] = load_frag16(&Bt[cur][(wave_n * 64 + j * 16) * STB], STB, 0, lane);
#pragma unroll
        for (int i = 0; i < 2; ++i)
#pragma unroll
            for (int j = 0; j < 4; ++j)
                acc[i][j] = wmma_bf16(afr[i], bfr[j], acc[i][j]);

        // ---- scatter B(next) transposed into the idle buffer ----
        if (has_next) {
#pragma unroll
            for (int i = 0; i < 2; ++i)
#pragma unroll
                for (int q = 0; q < 8; ++q)
                    Bt[nxt][(b_c8[i] + q) * STB + b_kr[i]] = breg[i].s[q];
            wait_async0();
        }
        __syncthreads();
    }

    // ---- epilogue (WMMA C layout: lane l, slot r -> row r+8*(l>>4), col l&15)
    const int rbase = 8 * ((lane >> 4) & 1);
    const int cloc  = lane & 15;
#pragma unroll
    for (int i = 0; i < 2; ++i)
#pragma unroll
        for (int j = 0; j < 4; ++j)
#pragma unroll
            for (int r = 0; r < 8; ++r) {
                int row = m0 + wave_m * 32 + i * 16 + r + rbase;
                int col = n0 + wave_n * 64 + j * 16 + cloc;
                float v = acc[i][j][r];
                if (bias)  v += bias[col];
                if (resid) v += resid[(size_t)row * N + col];
                if (OUT_BF16)
                    ((unsigned short*)out)[(size_t)row * N + col] = f32_to_bf16(v);
                else
                    ((float*)out)[(size_t)row * N + col] = v;
            }
}

// ---------------------------------------------------------------------------
// Flash attention.  grid = (S/64, NH, B), 128 threads = 4 waves.
// Each wave owns one 16-row Q tile; the block streams 64-key blocks.
// K tile staged via async global->LDS (row-major, directly usable as
// B fragments); V staged transposed through registers.
// QKV layout: [B*S, 6144], head h at col h*384 (+0 q, +128 k, +256 v).
// ---------------------------------------------------------------------------
#define STK 136   // K tile row stride (128 + 8); 272B, 16B-aligned
#define STV 72    // Vt/Pt row stride  (64 + 8); 144B, 16B-aligned

__global__ __launch_bounds__(128)
void flash_attn_kernel(const unsigned short* __restrict__ qkv,
                       const float* __restrict__ alibi,
                       const unsigned char* __restrict__ amask,
                       unsigned short* __restrict__ ctx) {
    __shared__ unsigned short Ks[64 * STK];       // [key][dim]
    __shared__ unsigned short Vt[128 * STV];      // [dim][key]
    __shared__ unsigned short Pt[4 * 16 * STV];   // per-wave [qrow][key]

    const int qblk = blockIdx.x;
    const int h    = blockIdx.y;
    const int b    = blockIdx.z;
    const int lane = threadIdx.x & 31;
    const int wave = threadIdx.x >> 5;
    const int qbase = qblk * 64 + wave * 16;
    const int rbase = 8 * ((lane >> 4) & 1);
    const int cloc  = lane & 15;

    // Q fragments straight from global (row stride 6144 elems = 12288B).
    const unsigned short* qrow = qkv + (size_t)(b * S_ + qbase) * H3_ + h * 384;
    v16bf qfr[4];
#pragma unroll
    for (int ks = 0; ks < 4; ++ks) qfr[ks] = load_frag16(qrow, H3_, ks * 32, lane);

    v8f o[8];
#pragma unroll
    for (int d = 0; d < 8; ++d) o[d] = (v8f)0.0f;
    float mrow[8], lrow[8];
#pragma unroll
    for (int r = 0; r < 8; ++r) { mrow[r] = -3.0e38f; lrow[r] = 0.0f; }

    const float scale = 0.08838834764831845f;   // 1/sqrt(128)
    const float* alibi_bh = alibi + (size_t)(b * NH_ + h) * S_;
    const unsigned char* mask_b = amask + (size_t)b * S_;

    const int nblocks = qblk + 1;               // causal
    for (int j = 0; j < nblocks; ++j) {
        const int kbase = j * 64;
        __syncthreads();
        // ---- stage K (async, row-major) and V (transposed via regs) ----
#pragma unroll
        for (int i = 0; i < 8; ++i) {
            int ch  = threadIdx.x + i * 128;    // 1024 chunks of 8 elems
            int row = ch >> 4;                  // 16 chunks/row
            int c8  = (ch & 15) * 8;
            const unsigned short* src =
                qkv + (size_t)(b * S_ + kbase + row) * H3_ + h * 384;
            async_copy_b128(lds_offset(&Ks[row * STK + c8]), src + HD_ + c8);
            union { uint4 v; unsigned short s[8]; } u;
            u.v = *(const uint4*)(src + 2 * HD_ + c8);
#pragma unroll
            for (int q = 0; q < 8; ++q) Vt[(c8 + q) * STV + row] = u.s[q];
        }
        wait_async0();
        __syncthreads();

        // ---- S = Q * K^T  (Ks rows are keys -> direct B fragments) ----
        v8f sacc[4];
#pragma unroll
        for (int nt = 0; nt < 4; ++nt) {
            sacc[nt] = (v8f)0.0f;
#pragma unroll
            for (int ks = 0; ks < 4; ++ks) {
                v16bf kf = load_frag16(&Ks[nt * 16 * STK], STK, ks * 32, lane);
                sacc[nt] = wmma_bf16(qfr[ks], kf, sacc[nt]);
            }
        }

        // ---- scale + alibi + causal/padding mask ----
#pragma unroll
        for (int nt = 0; nt < 4; ++nt)
#pragma unroll
            for (int r = 0; r < 8; ++r) {
                int col  = kbase + nt * 16 + cloc;
                int row  = qbase + r + rbase;
                float sv = sacc[nt][r] * scale + alibi_bh[col];
                bool ok  = (col <= row) && (mask_b[col] != 0);
                sacc[nt][r] = ok ? sv : -3.0e38f;
            }

        // ---- online softmax stats (row spread over 16 lanes of one slot) ----
        float alpha[8];
#pragma unroll
        for (int r = 0; r < 8; ++r) {
            float mx = sacc[0][r];
#pragma unroll
            for (int nt = 1; nt < 4; ++nt) mx = fmaxf(mx, sacc[nt][r]);
#pragma unroll
            for (int off = 1; off < 16; off <<= 1)
                mx = fmaxf(mx, __shfl_xor(mx, off, 32));
            float mnew = fmaxf(mrow[r], mx);
            alpha[r] = __expf(mrow[r] - mnew);
            mrow[r]  = mnew;
            float rs = 0.0f;
#pragma unroll
            for (int nt = 0; nt < 4; ++nt) {
                float p = __expf(sacc[nt][r] - mnew);
                sacc[nt][r] = p;
                rs += p;
            }
#pragma unroll
            for (int off = 1; off < 16; off <<= 1)
                rs += __shfl_xor(rs, off, 32);
            lrow[r] = lrow[r] * alpha[r] + rs;
        }
#pragma unroll
        for (int d = 0; d < 8; ++d)
#pragma unroll
            for (int r = 0; r < 8; ++r) o[d][r] *= alpha[r];

        // ---- P through LDS to re-layout C->A, then O += P * V ----
        unsigned short* pw = &Pt[wave * 16 * STV];
#pragma unroll
        for (int nt = 0; nt < 4; ++nt)
#pragma unroll
            for (int r = 0; r < 8; ++r)
                pw[(r + rbase) * STV + nt * 16 + cloc] = f32_to_bf16(sacc[nt][r]);
        __syncthreads();
#pragma unroll
        for (int kt = 0; kt < 2; ++kt) {
            v16bf pf = load_frag16(pw, STV, kt * 32, lane);
#pragma unroll
            for (int d = 0; d < 8; ++d) {
                v16bf vf = load_frag16(&Vt[d * 16 * STV], STV, kt * 32, lane);
                o[d] = wmma_bf16(pf, vf, o[d]);
            }
        }
    }

    // ---- normalize + write context (bf16, [B*S, H] with col = h*128+d) ----
#pragma unroll
    for (int d = 0; d < 8; ++d)
#pragma unroll
        for (int r = 0; r < 8; ++r) {
            int row = qbase + r + rbase;
            int col = d * 16 + cloc;
            float val = o[d][r] / lrow[r];
            ctx[(size_t)(b * S_ + row) * H_ + h * HD_ + col] = f32_to_bf16(val);
        }
}

// ---------------------------------------------------------------------------
// Host-side launcher
// ---------------------------------------------------------------------------
extern "C" void kernel_launch(void* const* d_in, const int* in_sizes, int n_in,
                              void* d_out, int out_size, void* d_ws, size_t ws_size,
                              hipStream_t stream) {
    (void)in_sizes; (void)n_in; (void)out_size; (void)ws_size;

    const float* hidden = (const float*)d_in[0];
    const float* resid  = (const float*)d_in[1];
    const float* alibi  = (const float*)d_in[2];
    const unsigned char* amask = (const unsigned char*)d_in[3];
    const float* W_qkv  = (const float*)d_in[4];
    const float* b_qkv  = (const float*)d_in[5];
    const float* W_out  = (const float*)d_in[6];
    const float* b_out  = (const float*)d_in[7];
    float* out = (float*)d_out;

    char* ws = (char*)d_ws;
    const size_t nX   = (size_t)B_ * S_ * H_;        // 8,388,608
    const size_t nWq  = (size_t)H_ * H3_;            // 12,582,912
    const size_t nWo  = (size_t)H_ * H_;             // 4,194,304
    const size_t nQKV = (size_t)B_ * S_ * H3_;       // 25,165,824

    unsigned short* Xb   = (unsigned short*)(ws);
    unsigned short* Wqb  = (unsigned short*)(ws + nX * 2);
    unsigned short* Wob  = (unsigned short*)(ws + (nX + nWq) * 2);
    unsigned short* QKVb = (unsigned short*)(ws + (nX + nWq + nWo) * 2);
    unsigned short* Ctxb = (unsigned short*)(ws + (nX + nWq + nWo + nQKV) * 2);

    // 1) fp32 -> bf16 conversions
    convert_bf16_kernel<<<2048, 256, 0, stream>>>(hidden, Xb,  nX);
    convert_bf16_kernel<<<2048, 256, 0, stream>>>(W_qkv,  Wqb, nWq);
    convert_bf16_kernel<<<2048, 256, 0, stream>>>(W_out,  Wob, nWo);

    // 2) fused QKV projection: [4096,2048] x [2048,6144] + b_qkv -> bf16
    {
        dim3 grid(H3_ / 128, (B_ * S_) / 128);
        gemm_bf16_kernel<true><<<grid, 256, 0, stream>>>(
            Xb, Wqb, b_qkv, nullptr, (void*)QKVb, B_ * S_, H3_, H_);
    }

    // 3) flash attention per (q-block, head, batch)
    {
        dim3 grid(S_ / 64, NH_, B_);
        flash_attn_kernel<<<grid, 128, 0, stream>>>(QKVb, alibi, amask, Ctxb);
    }

    // 4) output projection + b_out + residual -> fp32 d_out
    {
        dim3 grid(H_ / 128, (B_ * S_) / 128);
        gemm_bf16_kernel<false><<<grid, 256, 0, stream>>>(
            Ctxb, Wob, b_out, resid, (void*)out, B_ * S_, H_, H_);
    }
}
